// ContrastiveLoss_60722247631651
// MI455X (gfx1250) — matrix operand order repaired and checked by reading, not compiled, not run on hardware
//
#include <hip/hip_runtime.h>
#include <hip/hip_bf16.h>

typedef __attribute__((ext_vector_type(16))) _Float16 v16h;
typedef __attribute__((ext_vector_type(8)))  _Float16 v8h;
typedef __attribute__((ext_vector_type(4)))  _Float16 v4h;
typedef __attribute__((ext_vector_type(8)))  float    v8f;

#define N_PAIRS   4096
#define DIM       128
#define ZROWS     8192            // 2 * N_PAIRS
#define INV_TEMP  2.0f            // 1 / 0.5
#define EXP_SCALE 2.8853900817779268f   // INV_TEMP * log2(e)
#define ROWS_PER_WG 128
#define WAVES_PER_WG 8
#define NSPLIT    8               // column chunks
#define COLS_PER_CHUNK (ZROWS / NSPLIT)   // 1024

// ---------------------------------------------------------------------------
// Phase 1: row L2-normalize into f16 workspace. One wave per row, each lane
// owns 4 contiguous floats, wave32 butterfly reduction.
// ---------------------------------------------------------------------------
__global__ void nrm_kernel(const float* __restrict__ p1,
                           const float* __restrict__ p2,
                           _Float16* __restrict__ zh) {
  const int wave = threadIdx.x >> 5;
  const int lane = threadIdx.x & 31;
  const int row  = blockIdx.x * WAVES_PER_WG + wave;        // 0..8191
  const float* src = (row < N_PAIRS) ? (p1 + (size_t)row * DIM)
                                     : (p2 + (size_t)(row - N_PAIRS) * DIM);
  float4 v = ((const float4*)src)[lane];                    // 32*4 = 128
  float ss = v.x * v.x + v.y * v.y + v.z * v.z + v.w * v.w;
#pragma unroll
  for (int m = 16; m > 0; m >>= 1) ss += __shfl_xor(ss, m, 32);
  float inv = 1.0f / fmaxf(sqrtf(ss), 1e-12f);
  v4h h;
  h.x = (_Float16)(v.x * inv);
  h.y = (_Float16)(v.y * inv);
  h.z = (_Float16)(v.z * inv);
  h.w = (_Float16)(v.w * inv);
  ((v4h*)(zh + (size_t)row * DIM))[lane] = h;
}

// ---------------------------------------------------------------------------
// Fragment loaders for v_wmma_f32_16x16x32_f16 (ISA 7.12.2 layouts).
// ---------------------------------------------------------------------------
__device__ __forceinline__ v16h load_fragA(const _Float16* __restrict__ zh,
                                           int row_base, int kb, int m, int koff) {
  const _Float16* p = zh + (size_t)(row_base + m) * DIM + kb * 32 + koff;
  v8h lo = *(const v8h*)(p);        // K = kb*32 + koff + [0..7]
  v8h hi = *(const v8h*)(p + 16);   // K = kb*32 + 16 + koff + [0..7]
  return __builtin_shufflevector(lo, hi, 0, 1, 2, 3, 4, 5, 6, 7,
                                         8, 9, 10, 11, 12, 13, 14, 15);
}

__device__ __forceinline__ v16h load_fragB(const _Float16* __restrict__ zh,
                                           int col_base, int n, int kb, int hsel) {
  const _Float16* p = zh + (size_t)(col_base + n) * DIM + kb * 32 + hsel * 8;
  return __builtin_shufflevector(*(const v8h*)p, *(const v8h*)(p + 16), 0, 1, 2, 3, 4, 5, 6, 7, 8, 9, 10, 11, 12, 13, 14, 15);   // K = kb*32 + hsel*8 + [0..7], kb*32 + 16 + hsel*8 + [0..7]
}

struct BFrag { v16h b0, b1, b2, b3; };

__device__ __forceinline__ BFrag loadB4(const _Float16* __restrict__ zh,
                                        int j0, int n, int hsel) {
  BFrag f;
  f.b0 = load_fragB(zh, j0, n, 0, hsel);
  f.b1 = load_fragB(zh, j0, n, 1, hsel);
  f.b2 = load_fragB(zh, j0, n, 2, hsel);
  f.b3 = load_fragB(zh, j0, n, 3, hsel);
  return f;
}

// ---------------------------------------------------------------------------
// Phase 2: fused sim-GEMM + exp row-sum. Pure stream: no masks in the hot
// loop (diagonal + positive terms are reconstructed in phase 3).
// Grid: (64 row-blocks) x (8 col-chunks), 256 threads (8 waves). Each wave
// owns 16 rows and streams its 1024-column chunk in 16-wide tiles with
// register double-buffered B-fragment prefetch.
// ---------------------------------------------------------------------------
__global__ void __launch_bounds__(256)
ntxent_kernel(const _Float16* __restrict__ zh,
              float* __restrict__ rowsum_part) {   // [NSPLIT][ZROWS]
  const int lane  = threadIdx.x & 31;
  const int wave  = threadIdx.x >> 5;
  const int hsel  = lane >> 4;       // half-group: 0 or 1
  const int m     = lane & 15;       // A row / B col within tile
  const int koffA = hsel * 8;
  const int row0  = blockIdx.x * ROWS_PER_WG + wave * 16;
  const int cchunk = blockIdx.y;
  const int jstart = cchunk * COLS_PER_CHUNK;
  const int jend   = jstart + COLS_PER_CHUNK;

  // A fragments for this wave's 16 rows, K = 0..127 (registers, loaded once)
  v16h a0 = load_fragA(zh, row0, 0, m, koffA);
  v16h a1 = load_fragA(zh, row0, 1, m, koffA);
  v16h a2 = load_fragA(zh, row0, 2, m, koffA);
  v16h a3 = load_fragA(zh, row0, 3, m, koffA);

  float rowsum[8];
#pragma unroll
  for (int r = 0; r < 8; ++r) rowsum[r] = 0.0f;

  auto tile_compute = [&](const BFrag& f) {
    // two independent accumulator chains to hide WMMA RAW latency
    v8f c0 = {};
    v8f c1 = {};
    c0 = __builtin_amdgcn_wmma_f32_16x16x32_f16(false, a0, false, f.b0,
                                                (short)0, c0, false, false);
    c1 = __builtin_amdgcn_wmma_f32_16x16x32_f16(false, a1, false, f.b1,
                                                (short)0, c1, false, false);
    c0 = __builtin_amdgcn_wmma_f32_16x16x32_f16(false, a2, false, f.b2,
                                                (short)0, c0, false, false);
    c1 = __builtin_amdgcn_wmma_f32_16x16x32_f16(false, a3, false, f.b3,
                                                (short)0, c1, false, false);
#pragma unroll
    for (int r = 0; r < 8; ++r)
      rowsum[r] += __builtin_amdgcn_exp2f((c0[r] + c1[r]) * EXP_SCALE);
  };

  // software pipeline: prefetch next tile's B while computing current
  BFrag cur = loadB4(zh, jstart, m, hsel);
  for (int j0 = jstart; j0 < jend - 16; j0 += 16) {
    BFrag nxt = loadB4(zh, j0 + 16, m, hsel);
    tile_compute(cur);
    cur = nxt;
  }
  tile_compute(cur);

  // Reduce across the 16 lanes of each half-group (xor<16 stays in-group).
#pragma unroll
  for (int r = 0; r < 8; ++r) {
    float rs = rowsum[r];
#pragma unroll
    for (int msk = 1; msk < 16; msk <<= 1) rs += __shfl_xor(rs, msk, 32);
    rowsum[r] = rs;
  }

  if ((lane & 15) == 0) {            // lanes 0 and 16: 8 rows each
    const int base = row0 + hsel * 8;
#pragma unroll
    for (int r = 0; r < 8; ++r)
      rowsum_part[(size_t)cchunk * ZROWS + base + r] = rowsum[r];
  }
}

// ---------------------------------------------------------------------------
// Phase 3a: per-row loss. denominator = sum(partials) - exp(self/T); the
// self-dot and positive-dot are recomputed in f32 from the same f16 data the
// WMMA consumed (matches to ~ulp). One thread per row, plain store.
// ---------------------------------------------------------------------------
__global__ void __launch_bounds__(256)
rowloss_kernel(const _Float16* __restrict__ zh,
               const float* __restrict__ rowsum_part,
               float* __restrict__ rowloss) {
  const int row = blockIdx.x * 256 + threadIdx.x;     // 0..8191
  float rs = 0.0f;
#pragma unroll
  for (int c = 0; c < NSPLIT; ++c) rs += rowsum_part[(size_t)c * ZROWS + row];

  const _Float16* zi = zh + (size_t)row * DIM;
  const _Float16* zp = zh + (size_t)(row ^ N_PAIRS) * DIM;
  float self = 0.0f, pos = 0.0f;
#pragma unroll 8
  for (int k = 0; k < DIM; ++k) {
    const float x = (float)zi[k];
    self = fmaf(x, x, self);
    pos  = fmaf(x, (float)zp[k], pos);
  }
  rs -= __builtin_amdgcn_exp2f(self * EXP_SCALE);     // remove diagonal term
  rowloss[row] = __logf(rs) - pos * INV_TEMP;
}

// ---------------------------------------------------------------------------
// Phase 3b: fixed-order tree reduction of 8192 row losses -> scalar mean.
// ---------------------------------------------------------------------------
__global__ void __launch_bounds__(256)
reduce_kernel(const float* __restrict__ rowloss, float* __restrict__ out) {
  __shared__ float red[256];
  float acc = 0.0f;
  for (int row = threadIdx.x; row < ZROWS; row += 256) acc += rowloss[row];
  red[threadIdx.x] = acc;
  __syncthreads();
#pragma unroll
  for (int s = 128; s > 0; s >>= 1) {
    if (threadIdx.x < s) red[threadIdx.x] += red[threadIdx.x + s];
    __syncthreads();
  }
  if (threadIdx.x == 0) out[0] = red[0] * (1.0f / (float)ZROWS);
}

// ---------------------------------------------------------------------------
extern "C" void kernel_launch(void* const* d_in, const int* in_sizes, int n_in,
                              void* d_out, int out_size, void* d_ws, size_t ws_size,
                              hipStream_t stream) {
  const float* p1 = (const float*)d_in[0];
  const float* p2 = (const float*)d_in[1];
  float* out = (float*)d_out;

  _Float16* zh = (_Float16*)d_ws;                               // 2 MB
  float* rowsum_part = (float*)((char*)d_ws + (size_t)ZROWS * DIM * sizeof(_Float16));
  float* rowloss = rowsum_part + (size_t)NSPLIT * ZROWS;        // +256KB, +32KB

  nrm_kernel<<<ZROWS / WAVES_PER_WG, 256, 0, stream>>>(p1, p2, zh);
  dim3 grid2(ZROWS / ROWS_PER_WG, NSPLIT);
  ntxent_kernel<<<grid2, 256, 0, stream>>>(zh, rowsum_part);
  rowloss_kernel<<<ZROWS / 256, 256, 0, stream>>>(zh, rowsum_part, rowloss);
  reduce_kernel<<<1, 256, 0, stream>>>(rowloss, out);
}
